// MambaLayer_71880572666357
// MI455X (gfx1250) — compile-verified, matrix operand order checked by reading, not run on hardware
//
#include <hip/hip_runtime.h>
#include <hip/hip_bf16.h>
#include <math.h>

// ---------------- problem constants ----------------
#define DIMC   96
#define L_SEQ  8192          // T*H*W = 8*32*32
#define B_SZ   2
#define BL     16384         // B*L
#define DI     192           // D_INNER
#define DTR    6             // DT_RANK
#define DST    16            // D_STATE
#define GPAD   48            // DT_RANK + 2*D_STATE = 38, padded to 48
#define NXZ    384           // 2*D_INNER

typedef __attribute__((ext_vector_type(16))) _Float16 v16h;
typedef __attribute__((ext_vector_type(8)))  _Float16 v8h;
typedef __attribute__((ext_vector_type(8)))  float    v8f;

// serpentine spatial index of flat sequence position l (involution)
__device__ __forceinline__ int spatial_of(int l) {
    int t   = l >> 10;          // / (H*W)
    int rem = l & 1023;
    int h   = rem >> 5;         // / W
    int j   = rem & 31;
    int wj  = (h & 1) ? (31 - j) : j;
    return (t << 10) + (h << 5) + wj;
}

__device__ __forceinline__ float silu(float v) { return v / (1.f + expf(-v)); }

// ---------------- weight prep: f32 -> f16 (+pad), A = -exp(A_log) --------------
__global__ void k_prep(const float* __restrict__ w_in,  const float* __restrict__ w_xf,
                       const float* __restrict__ w_xb,  const float* __restrict__ w_out,
                       const float* __restrict__ Alog,  const float* __restrict__ Alogb,
                       _Float16* __restrict__ h_in, _Float16* __restrict__ h_xf,
                       _Float16* __restrict__ h_xb, _Float16* __restrict__ h_out,
                       float* __restrict__ A_f, float* __restrict__ A_b) {
    int t = blockIdx.x * 256 + threadIdx.x;            // 0 .. 36863
    if (t < NXZ * DIMC)  h_in[t]  = (_Float16)w_in[t];               // 384x96
    if (t < GPAD * DI) {                                             // 48x192 (pad rows >=38)
        int row = t / DI, col = t % DI;
        h_xf[t] = (row < (DTR + 2*DST)) ? (_Float16)w_xf[row*DI + col] : (_Float16)0.f;
        h_xb[t] = (row < (DTR + 2*DST)) ? (_Float16)w_xb[row*DI + col] : (_Float16)0.f;
    }
    if (t < DIMC * DI)   h_out[t] = (_Float16)w_out[t];              // 96x192
    if (t < DI * DST) { A_f[t] = -expf(Alog[t]); A_b[t] = -expf(Alogb[t]); }
}

// ---------------- serpentine gather + LayerNorm over 96 channels ----------------
// one wave (32 lanes) per token, 3 channels per lane
__global__ void k_ln(const float* __restrict__ x, const float* __restrict__ gamma,
                     const float* __restrict__ beta, float* __restrict__ xn) {
    int lane  = threadIdx.x & 31;
    int token = blockIdx.x * 4 + (threadIdx.x >> 5);   // < BL
    int b = token >> 13;                               // / L
    int l = token & (L_SEQ - 1);
    int sp = spatial_of(l);
    const float* base = x + (size_t)b * DIMC * L_SEQ + sp;
    float v[3], s = 0.f, s2 = 0.f;
    #pragma unroll
    for (int i = 0; i < 3; ++i) {
        int c = lane + 32 * i;
        float t = base[(size_t)c * L_SEQ];
        v[i] = t; s += t; s2 += t * t;
    }
    #pragma unroll
    for (int o = 16; o >= 1; o >>= 1) { s += __shfl_xor(s, o, 32); s2 += __shfl_xor(s2, o, 32); }
    float mu  = s  * (1.f / 96.f);
    float var = s2 * (1.f / 96.f) - mu * mu;
    float inv = rsqrtf(var + 1e-5f);
    float* out = xn + (size_t)token * DIMC;
    #pragma unroll
    for (int i = 0; i < 3; ++i) {
        int c = lane + 32 * i;
        out[c] = (v[i] - mu) * inv * gamma[c] + beta[c];
    }
}

// ---------------- WMMA GEMM: C(MxN) = A(MxK,f32) * W(NxK,f16)^T ----------------
// One wave computes a 16 x (16*TNT) strip: A loaded once per k-step (2x v8f = 64 B per lane),
// TNT back-to-back WMMAs against different weight tiles. K compile-time -> full
// unroll. A-element j of v16h holds k = kk + 16*(j>>3) + 8*half + (j&7), i.e.
// two contiguous 8-element runs per lane per k-step (vectorizable).
template <int K, int TNT>
__device__ __forceinline__ void gemm_strip(const float* __restrict__ Aact,
                                           const _Float16* __restrict__ Bw,
                                           int tm, int tile_n0, int lane, v8f acc[TNT]) {
    int half = lane >> 4, l16 = lane & 15;
    const float* arow = Aact + (size_t)(tm * 16 + l16) * K + 8 * half;
    #pragma unroll
    for (int t = 0; t < TNT; ++t) acc[t] = (v8f){};
    #pragma unroll
    for (int kk = 0; kk < K; kk += 32) {
        v8f af0 = *(const v8f*)(arow + kk);
        v8f af1 = *(const v8f*)(arow + kk + 16);
        v16h a;
        #pragma unroll
        for (int j = 0; j < 8; ++j) { a[j] = (_Float16)af0[j]; a[8 + j] = (_Float16)af1[j]; }
        #pragma unroll
        for (int t = 0; t < TNT; ++t) {
            const _Float16* brow = Bw + (size_t)((tile_n0 + t) * 16 + l16) * K + 8 * half;
            v8h b0 = *(const v8h*)(brow + kk);
            v8h b1 = *(const v8h*)(brow + kk + 16);
            v16h bm;
            #pragma unroll
            for (int j = 0; j < 8; ++j) { bm[j] = b0[j]; bm[8 + j] = b1[j]; }
            acc[t] = __builtin_amdgcn_wmma_f32_16x16x32_f16(false, a, false, bm,
                                                            (short)0, acc[t], false, false);
        }
    }
}

template <int K, int TNT>
__global__ void k_gemm_t(const float* __restrict__ Aact, const _Float16* __restrict__ Bw,
                         float* __restrict__ Cout, int groups_n, int N) {
    int lane  = threadIdx.x & 31;
    int wtile = blockIdx.x * 4 + (threadIdx.x >> 5);   // over (M/16)*groups_n
    int tm = wtile / groups_n, tg = wtile % groups_n;
    v8f acc[TNT];
    gemm_strip<K, TNT>(Aact, Bw, tm, tg * TNT, lane, acc);
    int half = lane >> 4, l16 = lane & 15;
    #pragma unroll
    for (int t = 0; t < TNT; ++t) {
        float* crow = Cout + (size_t)(tm * 16 + 8 * half) * N + (tg * TNT + t) * 16 + l16;
        #pragma unroll
        for (int v = 0; v < 8; ++v) crow[(size_t)v * N] = acc[t][v];
    }
}

// out_proj GEMM (K=192, N=96 in 6 tiles, TNT=3 -> 2 groups) fused with
// un-serpentine scatter straight into the (B,C,T,H,W) output tensor.
__global__ void k_gemm_out(const float* __restrict__ Aact, const _Float16* __restrict__ Bw,
                           float* __restrict__ out) {
    int lane  = threadIdx.x & 31;
    int wtile = blockIdx.x * 4 + (threadIdx.x >> 5);   // over (BL/16)*2
    int tm = wtile >> 1, tg = wtile & 1;
    v8f acc[3];
    gemm_strip<DI, 3>(Aact, Bw, tm, tg * 3, lane, acc);
    int half = lane >> 4, l16 = lane & 15;
    #pragma unroll
    for (int t = 0; t < 3; ++t) {
        int c = (tg * 3 + t) * 16 + l16;
        #pragma unroll
        for (int v = 0; v < 8; ++v) {
            int token = tm * 16 + 8 * half + v;
            int b = token >> 13;
            int l = token & (L_SEQ - 1);
            out[(size_t)(b * DIMC + c) * L_SEQ + spatial_of(l)] = acc[t][v];
        }
    }
}

// ---------------- depthwise causal conv1d (k=4) + SiLU; dir=1 reads flipped ----
__global__ void k_conv(const float* __restrict__ xz, const float* __restrict__ w,
                       const float* __restrict__ bias, float* __restrict__ xc, int dir) {
    int gid = blockIdx.x * 256 + threadIdx.x;          // < BL*DI
    int d  = gid % DI;
    int bl = gid / DI;
    int l  = bl & (L_SEQ - 1);
    int b  = bl >> 13;
    float acc = bias[d];
    #pragma unroll
    for (int k = 0; k < 4; ++k) {
        int li = l - 3 + k;
        if (li >= 0) {
            int seq = dir ? (L_SEQ - 1 - li) : li;
            acc += w[d * 4 + k] * xz[(size_t)(b * L_SEQ + seq) * NXZ + d];
        }
    }
    xc[(size_t)bl * DI + d] = silu(acc);
}

// ---------------- selective scan (sequential over L), dt-proj fused ----------------
// one thread per (b,d); h-state (16) and A row in registers.
__global__ void k_scan(const float* __restrict__ xc,  const float* __restrict__ dbl,
                       const float* __restrict__ xz,  const float* __restrict__ Aneg,
                       const float* __restrict__ wdt, const float* __restrict__ bdt,
                       const float* __restrict__ Dp,  float* __restrict__ ycomb, int dir) {
    int d = threadIdx.x;                               // 0..191
    int b = blockIdx.x;                                // 0..1
    float A[DST], h[DST], wr[DTR];
    #pragma unroll
    for (int n = 0; n < DST; ++n) { A[n] = Aneg[d * DST + n]; h[n] = 0.f; }
    #pragma unroll
    for (int r = 0; r < DTR; ++r) wr[r] = wdt[d * DTR + r];
    float bd = bdt[d], Dd = Dp[d];
    for (int l = 0; l < L_SEQ; ++l) {
        const float* row = dbl + (size_t)(b * L_SEQ + l) * GPAD;
        float dtr = bd;
        #pragma unroll
        for (int r = 0; r < DTR; ++r) dtr += row[r] * wr[r];
        float dt = (dtr > 20.f) ? dtr : log1pf(expf(dtr));   // softplus
        float x_t = xc[(size_t)(b * L_SEQ + l) * DI + d];
        float dtx = dt * x_t;
        float y = 0.f;
        #pragma unroll
        for (int n = 0; n < DST; ++n) {
            h[n] = expf(dt * A[n]) * h[n] + dtx * row[DTR + n];
            y += h[n] * row[DTR + DST + n];
        }
        y += Dd * x_t;
        int p = dir ? (L_SEQ - 1 - l) : l;
        float zv = xz[(size_t)(b * L_SEQ + p) * NXZ + DI + d];
        y *= silu(zv);
        float* o = ycomb + (size_t)(b * L_SEQ + p) * DI + d;
        if (dir) *o += y; else *o = y;
    }
}

// ---------------- host-side orchestration ----------------
extern "C" void kernel_launch(void* const* d_in, const int* in_sizes, int n_in,
                              void* d_out, int out_size, void* d_ws, size_t ws_size,
                              hipStream_t stream) {
    const float* x        = (const float*)d_in[0];
    const float* ln_g     = (const float*)d_in[1];
    const float* ln_b     = (const float*)d_in[2];
    const float* in_w     = (const float*)d_in[3];
    const float* conv_w   = (const float*)d_in[4];
    const float* conv_b   = (const float*)d_in[5];
    const float* xp_w     = (const float*)d_in[6];
    const float* dt_w     = (const float*)d_in[7];
    const float* dt_b     = (const float*)d_in[8];
    const float* A_log    = (const float*)d_in[9];
    const float* Dp       = (const float*)d_in[10];
    const float* conv_w_b = (const float*)d_in[11];
    const float* conv_b_b = (const float*)d_in[12];
    const float* xp_w_b   = (const float*)d_in[13];
    const float* dt_w_b   = (const float*)d_in[14];
    const float* dt_b_b   = (const float*)d_in[15];
    const float* A_log_b  = (const float*)d_in[16];
    const float* D_b      = (const float*)d_in[17];
    const float* out_w    = (const float*)d_in[18];
    float* out = (float*)d_out;

    // workspace carve-up (floats first, then f16 weights)
    float* ws    = (float*)d_ws;
    float* xn    = ws;                         // BL*96
    float* xz    = xn    + (size_t)BL * DIMC;  // BL*384
    float* xc_f  = xz    + (size_t)BL * NXZ;   // BL*192
    float* xc_b  = xc_f  + (size_t)BL * DI;    // BL*192
    float* dbl_f = xc_b  + (size_t)BL * DI;    // BL*48
    float* dbl_b = dbl_f + (size_t)BL * GPAD;  // BL*48
    float* ycomb = dbl_b + (size_t)BL * GPAD;  // BL*192
    float* A_f   = ycomb + (size_t)BL * DI;    // 3072
    float* A_b   = A_f + DI * DST;             // 3072
    _Float16* h_in  = (_Float16*)(A_b + DI * DST);   // 384*96
    _Float16* h_xf  = h_in  + NXZ * DIMC;            // 48*192
    _Float16* h_xb  = h_xf  + GPAD * DI;             // 48*192
    _Float16* h_out = h_xb  + GPAD * DI;             // 96*192

    // 1. weight prep
    k_prep<<<144, 256, 0, stream>>>(in_w, xp_w, xp_w_b, out_w, A_log, A_log_b,
                                    h_in, h_xf, h_xb, h_out, A_f, A_b);
    // 2. serpentine + layernorm  (BL tokens, 4 waves/block)
    k_ln<<<BL / 4, 128, 0, stream>>>(x, ln_g, ln_b, xn);
    // 3. in_proj: xz(BL,384) = xn(BL,96) * Win^T ; K=96, TNT=4 -> 6 groups
    k_gemm_t<DIMC, 4><<<(BL / 16) * 6 / 4, 128, 0, stream>>>(xn, h_in, xz, 6, NXZ);
    // 4. depthwise conv + silu, both directions
    k_conv<<<(BL * DI) / 256, 256, 0, stream>>>(xz, conv_w,   conv_b,   xc_f, 0);
    k_conv<<<(BL * DI) / 256, 256, 0, stream>>>(xz, conv_w_b, conv_b_b, xc_b, 1);
    // 5. x_proj: dbl(BL,48) = xc(BL,192) * Wx^T ; K=192, TNT=3 -> 1 group
    k_gemm_t<DI, 3><<<(BL / 16) * 1 / 4, 128, 0, stream>>>(xc_f, h_xf, dbl_f, 1, GPAD);
    k_gemm_t<DI, 3><<<(BL / 16) * 1 / 4, 128, 0, stream>>>(xc_b, h_xb, dbl_b, 1, GPAD);
    // 6. selective scans (forward writes, backward accumulates into flipped pos)
    k_scan<<<B_SZ, DI, 0, stream>>>(xc_f, dbl_f, xz, A_f, dt_w,   dt_b,   Dp,  ycomb, 0);
    k_scan<<<B_SZ, DI, 0, stream>>>(xc_b, dbl_b, xz, A_b, dt_w_b, dt_b_b, D_b, ycomb, 1);
    // 7. out_proj + un-serpentine scatter ; K=192, TNT=3 -> 2 groups
    k_gemm_out<<<(BL / 16) * 2 / 4, 128, 0, stream>>>(ycomb, h_out, out);
}
